// InteractionLayer_80942953660429
// MI455X (gfx1250) — matrix operand order repaired and checked by reading, not compiled, not run on hardware
//
#include <hip/hip_runtime.h>
#include <hip/hip_bf16.h>
#include <stdint.h>

typedef __bf16 bf16_t;
typedef __attribute__((ext_vector_type(16))) __bf16 v16bf;
typedef __attribute__((ext_vector_type(8)))  __bf16 v8bf;
typedef __attribute__((ext_vector_type(8)))  float  v8f;
typedef __attribute__((ext_vector_type(4)))  float  v4f;

#define BB 4
#define NN 4096
#define DD 128
#define SOFTMAX_SCALE 0.08838834764831845f   // 128^-0.5, folded into Q

#define KTILE_BYTES  (32 * 128 * 2)   // 8 KB per K tile
#define VTILE_BYTES  (128 * 32 * 2)   // 8 KB per Vt tile

static __device__ __forceinline__ v16bf cat8(v8bf lo, v8bf hi) {
  v16bf r;
#pragma unroll
  for (int i = 0; i < 8; ++i) { r[i] = lo[i]; r[i + 8] = hi[i]; }
  return r;
}

// ---------------------------------------------------------------------------
// Kernel 1: Q/K/V projection. One wave computes a 16-row tile of q,k,v.
// q,k stored row-major bf16; v stored transposed (Vt[b][d][n]) so the
// attention kernel's PV B-operand is a contiguous LDS row read.
// ---------------------------------------------------------------------------
__global__ __launch_bounds__(128)
void qkv_proj_kernel(const float* __restrict__ x,
                     const float* __restrict__ Wq, const float* __restrict__ bq,
                     const float* __restrict__ Wk, const float* __restrict__ bk,
                     const float* __restrict__ Wv, const float* __restrict__ bv,
                     bf16_t* __restrict__ qws, bf16_t* __restrict__ kws,
                     bf16_t* __restrict__ vtws)
{
  const int wave = threadIdx.x >> 5;
  const int lane = threadIdx.x & 31;
  const int half = lane >> 4;
  const int lm   = lane & 15;
  const int tile = blockIdx.x * 4 + wave;   // 0..1023 tiles of 16 rows
  const int row0 = tile * 16;               // flat row in [B*N]
  const int b    = row0 / NN;
  const int n0   = row0 % NN;

  // X tile as 4 A-fragments (K-chunks of 32). A layout: lane holds row lm,
  // K = c*32 + 8*half + {0..7} (elems 0..7) and +16 (elems 8..15).
  v16bf afrag[4];
#pragma unroll
  for (int c = 0; c < 4; ++c) {
    const float* p = x + (size_t)(row0 + lm) * DD + c * 32 + 8 * half;
    v4f f0 = *(const v4f*)(p);
    v4f f1 = *(const v4f*)(p + 4);
    v4f f2 = *(const v4f*)(p + 16);
    v4f f3 = *(const v4f*)(p + 20);
    v16bf a;
#pragma unroll
    for (int i = 0; i < 4; ++i) {
      a[i]      = (__bf16)f0[i];
      a[4 + i]  = (__bf16)f1[i];
      a[8 + i]  = (__bf16)f2[i];
      a[12 + i] = (__bf16)f3[i];
    }
    afrag[c] = a;
  }

  const float* Ws[3] = {Wq, Wk, Wv};
  const float* bs[3] = {bq, bk, bv};

#pragma unroll
  for (int m = 0; m < 3; ++m) {
    const float* W = Ws[m];
#pragma unroll
    for (int nt = 0; nt < 8; ++nt) {       // 8 output-column tiles of 16
      v8f acc = {};
#pragma unroll
      for (int c = 0; c < 4; ++c) {
        // B operand = W^T chunk: B[dd][o] = W[nt*16+o][c*32+dd].
        // B layout: lane holds column o=lm, K = 8*half + {0..7} (elems 0..7) and +16 (elems 8..15).
        const float* wp = W + (size_t)(nt * 16 + lm) * DD + c * 32 + 8 * half;
        v4f w0 = *(const v4f*)(wp);
        v4f w1 = *(const v4f*)(wp + 4);
        v4f w2 = *(const v4f*)(wp + 16);
        v4f w3 = *(const v4f*)(wp + 20);
        v16bf bf;
#pragma unroll
        for (int i = 0; i < 4; ++i) {
          bf[i]      = (__bf16)w0[i];
          bf[4 + i]  = (__bf16)w1[i];
          bf[8 + i]  = (__bf16)w2[i];
          bf[12 + i] = (__bf16)w3[i];
        }
        acc = __builtin_amdgcn_wmma_f32_16x16x32_bf16(
                  false, afrag[c], false, bf, (short)0, acc, false, false);
      }
      const float bias = bs[m][nt * 16 + lm];
      if (m == 0) {
#pragma unroll
        for (int r = 0; r < 8; ++r)
          qws[(size_t)(row0 + r + 8 * half) * DD + nt * 16 + lm] =
              (__bf16)((acc[r] + bias) * SOFTMAX_SCALE);
      } else if (m == 1) {
#pragma unroll
        for (int r = 0; r < 8; ++r)
          kws[(size_t)(row0 + r + 8 * half) * DD + nt * 16 + lm] =
              (__bf16)(acc[r] + bias);
      } else {
#pragma unroll
        for (int r = 0; r < 8; ++r)
          vtws[((size_t)b * DD + nt * 16 + lm) * NN + n0 + r + 8 * half] =
              (__bf16)(acc[r] + bias);
      }
    }
  }
}

// ---------------------------------------------------------------------------
// Kernel 2: flash attention with DOUBLE-BUFFERED async-to-LDS staging.
// Block = 4 waves, each owns a 16-query tile of the same batch. While the
// WMMAs chew on key-block kb, the async engine fills the other LDS buffer
// with kb+1 (ASYNCcnt completes in order per wave, so s_wait_asynccnt 8
// releases exactly the older buffer's 8 loads).
// ---------------------------------------------------------------------------
__global__ __launch_bounds__(128)
void flash_attn_kernel(const bf16_t* __restrict__ qw,
                       const bf16_t* __restrict__ kw,
                       const bf16_t* __restrict__ vtw,
                       float* __restrict__ out)
{
  __shared__ bf16_t k_lds[2][32 * 128];    // [buf][key][d]  2 x 8 KB
  __shared__ bf16_t vt_lds[2][128 * 32];   // [buf][d][key]  2 x 8 KB
  __shared__ bf16_t p_lds[4][16 * 32];     // per-wave P     4 x 1 KB

  const int tid  = threadIdx.x;
  const int wave = tid >> 5;
  const int lane = tid & 31;
  const int half = lane >> 4;
  const int lm   = lane & 15;

  const int blocksPerBatch = (NN / 16) / 4;              // 64
  const int b  = blockIdx.x / blocksPerBatch;
  const int q0 = ((blockIdx.x % blocksPerBatch) * 4 + wave) * 16;

  // Q A-fragments (softmax scale already folded in).
  v16bf qfrag[4];
#pragma unroll
  for (int c = 0; c < 4; ++c) {
    const bf16_t* p = qw + (size_t)(b * NN + q0 + lm) * DD + c * 32 + 8 * half;
    qfrag[c] = cat8(*(const v8bf*)p, *(const v8bf*)(p + 16));
  }

  v8f o[8];
#pragma unroll
  for (int nt = 0; nt < 8; ++nt) o[nt] = {};
  float mrow[8], lrow[8];
#pragma unroll
  for (int r = 0; r < 8; ++r) { mrow[r] = -3.0e38f; lrow[r] = 0.0f; }

  const uint32_t klds0  = (uint32_t)(uintptr_t)(&k_lds[0][0]);  // LDS off = low 32b of flat addr
  const uint32_t vtlds0 = (uint32_t)(uintptr_t)(&vt_lds[0][0]);
  const unsigned long long kbase  = (unsigned long long)(uintptr_t)kw;
  const unsigned long long vtbase = (unsigned long long)(uintptr_t)vtw;

  // Issue one key-block stage (8 async b128 loads per thread => 8 per wave).
  auto issue_stage = [&](int kb, int buf) {
    uint32_t gk = (uint32_t)((b * NN + kb * 32) * DD * 2) + tid * 64;
    uint32_t lk = klds0 + buf * KTILE_BYTES + tid * 64;
    uint32_t gv = (uint32_t)(((b * DD + tid) * NN + kb * 32) * 2);
    uint32_t lv = vtlds0 + buf * VTILE_BYTES + tid * 64;
#pragma unroll
    for (int i = 0; i < 4; ++i) {
      asm volatile("global_load_async_to_lds_b128 %0, %1, %2"
                   :: "v"(lk + i * 16), "v"(gk + i * 16), "s"(kbase) : "memory");
      asm volatile("global_load_async_to_lds_b128 %0, %1, %2"
                   :: "v"(lv + i * 16), "v"(gv + i * 16), "s"(vtbase) : "memory");
    }
  };

  const int NITER = NN / 32;   // 128 key blocks
  issue_stage(0, 0);

  for (int kb = 0; kb < NITER; ++kb) {
    const int cur = kb & 1;

    // Prefetch next block into the other buffer (its readers finished at the
    // end-of-iteration barrier of kb-1), then release the current buffer.
    if (kb + 1 < NITER) {
      issue_stage(kb + 1, cur ^ 1);
      asm volatile("s_wait_asynccnt 8" ::: "memory");   // oldest 8 (== cur buf) done
    } else {
      asm volatile("s_wait_asynccnt 0" ::: "memory");
    }
    __syncthreads();

    const bf16_t* kl = &k_lds[cur][0];
    const bf16_t* vl = &vt_lds[cur][0];

    // ---- S = Q K^T : two 16-key subtiles, K-dim 128 in 4 chunks
    v8f s0 = {}, s1 = {};
#pragma unroll
    for (int c = 0; c < 4; ++c) {
      const bf16_t* kp0 = kl + (lm)      * 128 + c * 32 + 8 * half;
      const bf16_t* kp1 = kl + (16 + lm) * 128 + c * 32 + 8 * half;
      v16bf b0 = cat8(*(const v8bf*)kp0, *(const v8bf*)(kp0 + 16));
      v16bf b1 = cat8(*(const v8bf*)kp1, *(const v8bf*)(kp1 + 16));
      s0 = __builtin_amdgcn_wmma_f32_16x16x32_bf16(false, qfrag[c], false, b0,
                                                   (short)0, s0, false, false);
      s1 = __builtin_amdgcn_wmma_f32_16x16x32_bf16(false, qfrag[c], false, b1,
                                                   (short)0, s1, false, false);
    }

    // ---- online softmax; C-frag row of VGPR r is (r + 8*half), col = lm
#pragma unroll
    for (int r = 0; r < 8; ++r) {
      float mr = fmaxf(s0[r], s1[r]);
#pragma unroll
      for (int d = 1; d < 16; d <<= 1) mr = fmaxf(mr, __shfl_xor(mr, d, 32));
      const float mn    = fmaxf(mrow[r], mr);
      const float alpha = __expf(mrow[r] - mn);
      const float p0 = __expf(s0[r] - mn);
      const float p1 = __expf(s1[r] - mn);
      float rs = p0 + p1;
#pragma unroll
      for (int d = 1; d < 16; d <<= 1) rs += __shfl_xor(rs, d, 32);
      lrow[r] = lrow[r] * alpha + rs;
      mrow[r] = mn;
#pragma unroll
      for (int nt = 0; nt < 8; ++nt) o[nt][r] *= alpha;
      p_lds[wave][(r + 8 * half) * 32 + lm]      = (__bf16)p0;
      p_lds[wave][(r + 8 * half) * 32 + 16 + lm] = (__bf16)p1;
    }

    // ---- re-layout P (C-frag -> A-frag) through per-wave LDS buffer
    const bf16_t* pp = &p_lds[wave][lm * 32 + 8 * half];
    v16bf pf = cat8(*(const v8bf*)pp, *(const v8bf*)(pp + 16));

    // ---- O += P (16x32) x V (32x16 per column tile); Vt rows are contiguous
#pragma unroll
    for (int nt = 0; nt < 8; ++nt) {
      const bf16_t* vp = vl + (nt * 16 + lm) * 32 + 8 * half;
      v16bf vf = cat8(*(const v8bf*)vp, *(const v8bf*)(vp + 16));
      o[nt] = __builtin_amdgcn_wmma_f32_16x16x32_bf16(false, pf, false, vf,
                                                      (short)0, o[nt], false, false);
    }

    __syncthreads();   // readers of buf `cur` done -> kb+2 may overwrite it
  }

  // ---- epilogue: O / l, store fp32
#pragma unroll
  for (int r = 0; r < 8; ++r) {
    const float inv = 1.0f / lrow[r];
#pragma unroll
    for (int nt = 0; nt < 8; ++nt)
      out[(size_t)(b * NN + q0 + r + 8 * half) * DD + nt * 16 + lm] = o[nt][r] * inv;
  }
}

// ---------------------------------------------------------------------------
extern "C" void kernel_launch(void* const* d_in, const int* in_sizes, int n_in,
                              void* d_out, int out_size, void* d_ws, size_t ws_size,
                              hipStream_t stream) {
  (void)in_sizes; (void)n_in; (void)out_size; (void)ws_size;
  const float* x  = (const float*)d_in[0];
  const float* Wq = (const float*)d_in[1];
  const float* bq = (const float*)d_in[2];
  const float* Wk = (const float*)d_in[3];
  const float* bk = (const float*)d_in[4];
  const float* Wv = (const float*)d_in[5];
  const float* bv = (const float*)d_in[6];
  float* out = (float*)d_out;

  bf16_t* qws  = (bf16_t*)d_ws;                     // 4 MB
  bf16_t* kws  = qws + (size_t)BB * NN * DD;        // 4 MB
  bf16_t* vtws = kws + (size_t)BB * NN * DD;        // 4 MB (V transposed)

  hipLaunchKernelGGL(qkv_proj_kernel, dim3((BB * NN / 16) / 4), dim3(128), 0, stream,
                     x, Wq, bq, Wk, bk, Wv, bv, qws, kws, vtws);
  hipLaunchKernelGGL(flash_attn_kernel, dim3(BB * (NN / 16) / 4), dim3(128), 0, stream,
                     qws, kws, vtws, out);
}
